// PopulationGCN_65987877535843
// MI455X (gfx1250) — matrix operand order repaired and checked by reading, not compiled, not run on hardware
//
#include <hip/hip_runtime.h>
#include <math.h>

#define NN   20000      // nodes
#define TT   20         // timesteps
#define NE   1280000    // edges
#define EMB  64
#define HID  128
#define G4H  512        // 4*HID gates
#define ODIM 10

typedef __attribute__((ext_vector_type(16))) _Float16 v16h;
typedef __attribute__((ext_vector_type(8)))  float    v8f;

union Frag16 { v16h v; _Float16 h[16]; uint4 q[2]; };

// ---------------------------------------------------------------------------
// WMMA GEMM: C[M x G] (f32) (+)= A[M x K] (f16, row-major, lda) x Bp
// (f16, pre-packed 32x16 K-major fragments of B[K x G]).
// One wave computes an 80(M) x 64(G) tile: 5 M-subtiles x 4 N-subtiles,
// 20 accumulators, K-loop double-buffered (nk always even: K in {64,128,256}).
// M must be a multiple of 80 (20000 = 80*250).
// ---------------------------------------------------------------------------
__device__ __forceinline__ void load_a5(Frag16 (&a)[5],
                                        const _Float16* const (&arow)[5], int kt)
{
    #pragma unroll
    for (int m = 0; m < 5; ++m) {
        const _Float16* ap = arow[m] + (kt << 5);
        a[m].q[0] = *(const uint4*)(ap);
        a[m].q[1] = *(const uint4*)(ap + 8);
    }
}
__device__ __forceinline__ void load_b4(Frag16 (&b)[4],
                                        const _Float16* blane, int kt, int gTiles16)
{
    #pragma unroll
    for (int j = 0; j < 4; ++j) {
        const _Float16* bp = blane + ((long)kt * gTiles16 + j) * 512;
        b[j].q[0] = *(const uint4*)(bp);
        b[j].q[1] = *(const uint4*)(bp + 8);
    }
}
__device__ __forceinline__ void mma20(v8f (&acc)[5][4], Frag16 (&a)[5], Frag16 (&b)[4])
{
    #pragma unroll
    for (int m = 0; m < 5; ++m)
        #pragma unroll
        for (int j = 0; j < 4; ++j)
            acc[m][j] = __builtin_amdgcn_wmma_f32_16x16x32_f16(
                false, a[m].v, false, b[j].v, (short)0, acc[m][j], false, false);
}

__global__ void __launch_bounds__(128)
gemm_f16_wmma(const _Float16* __restrict__ A, int lda,
              const _Float16* __restrict__ Bp,
              float* __restrict__ C,
              int M, int K, int G, int accum)
{
    const int lane     = threadIdx.x & 31;
    const int wave     = blockIdx.x * (blockDim.x >> 5) + (threadIdx.x >> 5);
    const int gTiles64 = G >> 6;
    const int mg       = wave / gTiles64;          // 80-row group
    const int gt       = wave % gTiles64;          // 64-col group
    if (mg * 80 >= M) return;                      // wave-uniform

    const int halfSel   = lane >> 4;               // 0: lanes 0-15, 1: lanes 16-31
    const int rowInTile = lane & 15;
    const int kA        = halfSel * 16;            // A frag K split {0..15}/{16..31}
    const int gTiles16  = G >> 4;

    // Hoisted per-subtile A row pointers and the per-lane B base.
    const _Float16* arow[5];
    #pragma unroll
    for (int m = 0; m < 5; ++m)
        arow[m] = A + ((long)(mg * 80 + m * 16) + rowInTile) * (long)lda + kA;
    const _Float16* blane = Bp + ((long)(gt << 2) * 32 + lane) * 16;

    v8f acc[5][4];
    #pragma unroll
    for (int m = 0; m < 5; ++m)
        #pragma unroll
        for (int j = 0; j < 4; ++j) {
            const int col = (gt << 6) + (j << 4) + rowInTile;
            #pragma unroll
            for (int r = 0; r < 8; ++r) {
                const long rr = (long)(mg * 80 + m * 16) + r + (halfSel << 3);
                acc[m][j][r] = accum ? C[rr * (long)G + col] : 0.0f;
            }
        }

    const int nk = K >> 5;                         // even: 2, 4 or 8
    Frag16 a0[5], a1[5], b0[4], b1[4];
    load_a5(a0, arow, 0);
    load_b4(b0, blane, 0, gTiles16);
    for (int kt = 0; kt < nk; kt += 2) {
        load_a5(a1, arow, kt + 1);                 // prefetch odd chunk
        load_b4(b1, blane, kt + 1, gTiles16);
        mma20(acc, a0, b0);                        // compute even chunk
        if (kt + 2 < nk) {                         // prefetch next even chunk
            load_a5(a0, arow, kt + 2);
            load_b4(b0, blane, kt + 2, gTiles16);
        }
        mma20(acc, a1, b1);                        // compute odd chunk
    }

    #pragma unroll
    for (int m = 0; m < 5; ++m)
        #pragma unroll
        for (int j = 0; j < 4; ++j) {
            const int col = (gt << 6) + (j << 4) + rowInTile;
            #pragma unroll
            for (int r = 0; r < 8; ++r) {
                const long rr = (long)(mg * 80 + m * 16) + r + (halfSel << 3);
                C[rr * (long)G + col] = acc[m][j][r];
            }
        }
}

// Pack B[K x G] into 32x16 f16 WMMA fragments in this file's contiguous k order (same as A):
// lanes 0-15 hold K=kt*32+e of column gt*16+lane; lanes 16-31 hold K=kt*32+16+e.
// srcTransposed=1: W stored [G x K] row-major (torch Wih/Whh); else [K x G].
__global__ void pack_weight_frag(const float* __restrict__ W,
                                 _Float16* __restrict__ Bp,
                                 int K, int G, int srcTransposed)
{
    long i = (long)blockIdx.x * blockDim.x + threadIdx.x;
    if (i >= (long)K * G) return;
    int  e    = (int)(i & 15);
    int  lane = (int)((i >> 4) & 31);
    long frag = i >> 9;
    int  ngt  = G >> 4;
    int  kt   = (int)(frag / ngt), gt = (int)(frag % ngt);
    int  col  = gt * 16 + (lane & 15);
    int  k    = kt * 32 + ((lane >> 4) << 4) + e;
    float w   = srcTransposed ? W[(long)col * K + k] : W[(long)k * G + col];
    Bp[i] = (_Float16)w;
}

// Embedding gather + f16 cast:  e16[t][n][j] = emb[x[n][t]][j]
__global__ void embed_gather(const int* __restrict__ x,
                             const float* __restrict__ emb,
                             _Float16* __restrict__ e16)
{
    long i = (long)blockIdx.x * blockDim.x + threadIdx.x;
    if (i >= (long)NN * TT * EMB) return;
    int  j  = (int)(i & 63);
    long nt = i >> 6;
    int  t  = (int)(nt % TT);
    long n  = nt / TT;
    int idx = x[n * TT + t];
    e16[((long)t * NN + n) * EMB + j] = (_Float16)emb[(long)idx * EMB + j];
}

__device__ __forceinline__ float sigmoidf_(float v) { return 1.0f / (1.0f + __expf(-v)); }

// LSTM pointwise gates (torch order i,f,g,o). gates already = x@Wih^T + h@Whh^T.
__global__ void lstm_pointwise(const float* __restrict__ gates,
                               const float* __restrict__ bih,
                               const float* __restrict__ bhh,
                               float* __restrict__ c,
                               _Float16* __restrict__ hOut, int ldh,
                               const int* __restrict__ lengths, int t,
                               _Float16* __restrict__ last, int first)
{
    long i = (long)blockIdx.x * blockDim.x + threadIdx.x;
    if (i >= (long)NN * HID) return;
    long n = i >> 7;
    int  j = (int)(i & 127);
    const float* g = gates + n * G4H;
    float ig = sigmoidf_(g[j]         + bih[j]         + bhh[j]);
    float fg = sigmoidf_(g[HID + j]   + bih[HID + j]   + bhh[HID + j]);
    float gg = tanhf    (g[2*HID + j] + bih[2*HID + j] + bhh[2*HID + j]);
    float og = sigmoidf_(g[3*HID + j] + bih[3*HID + j] + bhh[3*HID + j]);
    float cp = first ? 0.0f : c[i];
    float cn = fg * cp + ig * gg;
    float hn = og * tanhf(cn);
    c[i] = cn;
    _Float16 hh = (_Float16)hn;
    hOut[n * (long)ldh + j] = hh;
    if (last && (lengths[n] - 1 == t)) last[n * (long)(2 * HID) + j] = hh;
}

// --- GCN norm ---
__global__ void deg_init(float* deg)
{
    int n = blockIdx.x * blockDim.x + threadIdx.x;
    if (n < NN) deg[n] = 1.0f;                   // self-loop weight
}
__global__ void deg_accum(const int* __restrict__ dst, const float* __restrict__ w,
                          float* __restrict__ deg)
{
    long e = (long)blockIdx.x * blockDim.x + threadIdx.x;
    if (e < NE) atomicAdd(&deg[dst[e]], w[e]);
}
__global__ void deg_rsqrt(float* deg)
{
    int n = blockIdx.x * blockDim.x + threadIdx.x;
    if (n < NN) { float d = deg[n]; deg[n] = d > 0.0f ? rsqrtf(d) : 0.0f; }
}

// agg[dst] += dinv[src]*w*dinv[dst] * xw[src]
__global__ void gcn_scatter(const int* __restrict__ src, const int* __restrict__ dst,
                            const float* __restrict__ w, const float* __restrict__ dinv,
                            const float* __restrict__ xw, float* __restrict__ agg)
{
    long i = (long)blockIdx.x * blockDim.x + threadIdx.x;
    if (i >= (long)NE * HID) return;
    int  j = (int)(i & 127);
    long e = i >> 7;
    int  s = src[e], d = dst[e];
    float norm = dinv[s] * w[e] * dinv[d];
    atomicAdd(&agg[(long)d * HID + j], norm * xw[(long)s * HID + j]);
}
__global__ void gcn_self(const float* __restrict__ dinv, const float* __restrict__ xw,
                         float* __restrict__ agg)
{
    long i = (long)blockIdx.x * blockDim.x + threadIdx.x;
    if (i >= (long)NN * HID) return;
    long n = i >> 7;
    float dv = dinv[n];
    agg[i] += dv * dv * xw[i];
}
__global__ void bias_relu_f16(const float* __restrict__ agg, const float* __restrict__ b,
                              _Float16* __restrict__ h16)
{
    long i = (long)blockIdx.x * blockDim.x + threadIdx.x;
    if (i >= (long)NN * HID) return;
    int j = (int)(i & 127);
    float v = agg[i] + b[j];
    h16[i] = (_Float16)(v > 0.0f ? v : 0.0f);
}
// out = relu(agg + b2) @ Wfc + bfc  (ODIM=10, trivially small)
__global__ void fc_out(const float* __restrict__ agg, const float* __restrict__ b2,
                       const float* __restrict__ Wfc, const float* __restrict__ bfc,
                       float* __restrict__ out)
{
    long i = (long)blockIdx.x * blockDim.x + threadIdx.x;
    if (i >= (long)NN * ODIM) return;
    int  o = (int)(i % ODIM);
    long n = i / ODIM;
    const float* row = agg + n * HID;
    float s = bfc[o];
    for (int k = 0; k < HID; ++k) {
        float v = row[k] + b2[k];
        v = v > 0.0f ? v : 0.0f;
        s += v * Wfc[k * ODIM + o];
    }
    out[i] = s;
}

static inline int cdiv(long a, int b) { return (int)((a + b - 1) / b); }

extern "C" void kernel_launch(void* const* d_in, const int* in_sizes, int n_in,
                              void* d_out, int out_size, void* d_ws, size_t ws_size,
                              hipStream_t stream)
{
    (void)in_sizes; (void)n_in; (void)out_size; (void)ws_size;

    const int*   x       = (const int*)  d_in[0];
    const int*   lengths = (const int*)  d_in[1];
    const int*   eidx    = (const int*)  d_in[2];
    const float* ew      = (const float*)d_in[3];
    const float* emb     = (const float*)d_in[4];
    const float* Wih[4]  = { (const float*)d_in[5],  (const float*)d_in[9],
                             (const float*)d_in[13], (const float*)d_in[17] };
    const float* Whh[4]  = { (const float*)d_in[6],  (const float*)d_in[10],
                             (const float*)d_in[14], (const float*)d_in[18] };
    const float* bih[4]  = { (const float*)d_in[7],  (const float*)d_in[11],
                             (const float*)d_in[15], (const float*)d_in[19] };
    const float* bhh[4]  = { (const float*)d_in[8],  (const float*)d_in[12],
                             (const float*)d_in[16], (const float*)d_in[20] };
    const float* W1  = (const float*)d_in[21];
    const float* b1  = (const float*)d_in[22];
    const float* W2  = (const float*)d_in[23];
    const float* b2  = (const float*)d_in[24];
    const float* Wfc = (const float*)d_in[25];
    const float* bfc = (const float*)d_in[26];
    float* out = (float*)d_out;

    // -------- workspace carve (256B aligned chunks) --------
    char* p = (char*)d_ws;
    auto carve = [&](size_t bytes) -> char* {
        char* r = p; p += (bytes + 255) & ~(size_t)255; return r;
    };
    _Float16* e16   = (_Float16*)carve((size_t)TT * NN * EMB * 2);       //  51 MB
    _Float16* h0    = (_Float16*)carve((size_t)TT * NN * 2 * HID * 2);   // 205 MB
    float*    gates = (float*)   carve((size_t)NN * G4H * 4);            //  41 MB
    float*    cbuf  = (float*)   carve((size_t)NN * HID * 4);
    _Float16* h1cur = (_Float16*)carve((size_t)NN * HID * 2);
    _Float16* lastb = (_Float16*)carve((size_t)NN * 2 * HID * 2);
    float*    xw    = (float*)   carve((size_t)NN * HID * 4);
    float*    agg   = (float*)   carve((size_t)NN * HID * 4);
    _Float16* hg16  = (_Float16*)carve((size_t)NN * HID * 2);
    float*    dinv  = (float*)   carve((size_t)NN * 4);
    const int kin[4] = { EMB, EMB, 2 * HID, 2 * HID };
    _Float16 *pWih[4], *pWhh[4];
    for (int d = 0; d < 4; ++d) {
        pWih[d] = (_Float16*)carve((size_t)kin[d] * G4H * 2);
        pWhh[d] = (_Float16*)carve((size_t)HID * G4H * 2);
    }
    _Float16* pW1 = (_Float16*)carve((size_t)2 * HID * HID * 2);
    _Float16* pW2 = (_Float16*)carve((size_t)HID * HID * 2);

    const int TB = 256;
    auto gemm = [&](const _Float16* A, int lda, const _Float16* Bp, float* C,
                    int M, int K, int Gc, int accum) {
        int waves = (M / 80) * (Gc / 64);          // M=20000 -> 250 row-groups
        gemm_f16_wmma<<<waves / 4, 128, 0, stream>>>(A, lda, Bp, C, M, K, Gc, accum);
    };

    // -------- pack weights into WMMA fragment order --------
    for (int d = 0; d < 4; ++d) {
        pack_weight_frag<<<cdiv((long)kin[d] * G4H, TB), TB, 0, stream>>>(
            Wih[d], pWih[d], kin[d], G4H, 1);
        pack_weight_frag<<<cdiv((long)HID * G4H, TB), TB, 0, stream>>>(
            Whh[d], pWhh[d], HID, G4H, 1);
    }
    pack_weight_frag<<<cdiv((long)2 * HID * HID, TB), TB, 0, stream>>>(W1, pW1, 2 * HID, HID, 0);
    pack_weight_frag<<<cdiv((long)HID * HID, TB), TB, 0, stream>>>(W2, pW2, HID, HID, 0);

    // -------- embedding --------
    embed_gather<<<cdiv((long)NN * TT * EMB, TB), TB, 0, stream>>>(x, emb, e16);

    // -------- BiLSTM layer 0 (writes h0[t][n][dir*128 + j]) --------
    for (int dir = 0; dir < 2; ++dir) {
        for (int s = 0; s < TT; ++s) {
            int t = dir ? (TT - 1 - s) : s;
            gemm(e16 + (size_t)t * NN * EMB, EMB, pWih[dir], gates, NN, EMB, G4H, 0);
            if (s > 0) {
                int tp = dir ? (t + 1) : (t - 1);
                gemm(h0 + (size_t)tp * NN * 2 * HID + dir * HID, 2 * HID,
                     pWhh[dir], gates, NN, HID, G4H, 1);
            }
            lstm_pointwise<<<cdiv((long)NN * HID, TB), TB, 0, stream>>>(
                gates, bih[dir], bhh[dir], cbuf,
                h0 + (size_t)t * NN * 2 * HID + dir * HID, 2 * HID,
                lengths, t, (_Float16*)nullptr, s == 0);
        }
    }

    // -------- BiLSTM layer 1 (records last[n][dir*128+j] at t==len-1) --------
    for (int dir = 0; dir < 2; ++dir) {
        int pi = 2 + dir;
        for (int s = 0; s < TT; ++s) {
            int t = dir ? (TT - 1 - s) : s;
            gemm(h0 + (size_t)t * NN * 2 * HID, 2 * HID, pWih[pi], gates, NN, 2 * HID, G4H, 0);
            if (s > 0)
                gemm(h1cur, HID, pWhh[pi], gates, NN, HID, G4H, 1);
            lstm_pointwise<<<cdiv((long)NN * HID, TB), TB, 0, stream>>>(
                gates, bih[pi], bhh[pi], cbuf,
                h1cur, HID, lengths, t, lastb + dir * HID, s == 0);
        }
    }

    // -------- GCN norm --------
    const int* src = eidx;
    const int* dst = eidx + NE;
    deg_init <<<cdiv(NN, TB), TB, 0, stream>>>(dinv);
    deg_accum<<<cdiv(NE, TB), TB, 0, stream>>>(dst, ew, dinv);
    deg_rsqrt<<<cdiv(NN, TB), TB, 0, stream>>>(dinv);

    // -------- conv1: relu(scatter(last @ W1) + b1) --------
    gemm(lastb, 2 * HID, pW1, xw, NN, 2 * HID, HID, 0);
    hipMemsetAsync(agg, 0, (size_t)NN * HID * 4, stream);
    gcn_scatter<<<cdiv((long)NE * HID, TB), TB, 0, stream>>>(src, dst, ew, dinv, xw, agg);
    gcn_self   <<<cdiv((long)NN * HID, TB), TB, 0, stream>>>(dinv, xw, agg);
    bias_relu_f16<<<cdiv((long)NN * HID, TB), TB, 0, stream>>>(agg, b1, hg16);

    // -------- conv2 + FC --------
    gemm(hg16, HID, pW2, xw, NN, HID, HID, 0);
    hipMemsetAsync(agg, 0, (size_t)NN * HID * 4, stream);
    gcn_scatter<<<cdiv((long)NE * HID, TB), TB, 0, stream>>>(src, dst, ew, dinv, xw, agg);
    gcn_self   <<<cdiv((long)NN * HID, TB), TB, 0, stream>>>(dinv, xw, agg);
    fc_out     <<<cdiv((long)NN * ODIM, TB), TB, 0, stream>>>(agg, b2, Wfc, bfc, out);
}
